// SAGEConv_78580721648259
// MI455X (gfx1250) — compile-verified, matrix operand order checked by reading, not run on hardware
//
#include <hip/hip_runtime.h>

typedef __attribute__((ext_vector_type(16))) __bf16 v16bf;
typedef __attribute__((ext_vector_type(8)))  float  v8f;

#define N_NODES 65536
#define IN_DIM  512
#define OUT_DIM 512
#define K_DIM   1024   // 2 * IN_DIM
#define N_EDGES 524288
#define BK      128    // K-chunk staged in LDS per double-buffer slot
#define BSTR    136    // 128 + 8 ushort pad -> conflict-free ds_load_b128

// Async global->LDS staging (ASYNCcnt) via inline CDNA5 asm on the device pass;
// host pass (and non-AMDGCN) parses the synchronous fallback.
#if defined(__AMDGCN__)
#define USE_ASYNC_LDS 1
#else
#define USE_ASYNC_LDS 0
#endif

// ---- float -> bf16 round-to-nearest-even ----
__device__ __forceinline__ unsigned short f2bf(float x) {
  unsigned u = __float_as_uint(x);
  unsigned r = u + 0x7FFFu + ((u >> 16) & 1u);
  return (unsigned short)(r >> 16);
}

// ---- 0) zero the aggregation buffer (workspace is poisoned by harness) ----
__global__ __launch_bounds__(256) void zero_f32_kernel(float4* __restrict__ p, int n4) {
  int i = blockIdx.x * 256 + threadIdx.x;
  if (i < n4) p[i] = make_float4(0.f, 0.f, 0.f, 0.f);
}

// ---- 1) scatter-sum: one wave per edge, agg[src] += feat[tgt] ----
__global__ __launch_bounds__(256) void scatter_kernel(const float* __restrict__ feat,
                                                      const long long* __restrict__ ei,
                                                      float* __restrict__ agg) {
  unsigned e    = (blockIdx.x * 256u + threadIdx.x) >> 5;
  unsigned lane = threadIdx.x & 31u;
  if (e >= N_EDGES) return;
  long long src = ei[e];
  long long tgt = ei[(size_t)N_EDGES + e];
  const float4* f = (const float4*)(feat + (size_t)tgt * IN_DIM);
  float* a = agg + (size_t)src * IN_DIM;
#pragma unroll
  for (int i = 0; i < IN_DIM / 4 / 32; ++i) {      // 4 iterations
    int idx = (int)lane + i * 32;
    float4 v = f[idx];
    atomicAdd(a + idx * 4 + 0, v.x);
    atomicAdd(a + idx * 4 + 1, v.y);
    atomicAdd(a + idx * 4 + 2, v.z);
    atomicAdd(a + idx * 4 + 3, v.w);
  }
}

// ---- 2a) pack A = [feat | agg] as bf16 [N_NODES, 1024] ----
__global__ __launch_bounds__(256) void pack_A_kernel(const float* __restrict__ feat,
                                                     const float* __restrict__ agg,
                                                     unsigned short* __restrict__ A) {
  size_t t    = (size_t)blockIdx.x * 256 + threadIdx.x;
  size_t base = t * 8;                       // 8 elements per thread, never crosses halves
  size_t row  = base >> 10;
  int    k    = (int)(base & 1023);
  const float* src = (k < IN_DIM) ? (feat + row * IN_DIM + k)
                                  : (agg  + row * IN_DIM + (k - IN_DIM));
  float4 lo = ((const float4*)src)[0];
  float4 hi = ((const float4*)src)[1];
  alignas(16) unsigned short o[8] = { f2bf(lo.x), f2bf(lo.y), f2bf(lo.z), f2bf(lo.w),
                                      f2bf(hi.x), f2bf(hi.y), f2bf(hi.z), f2bf(hi.w) };
  *(uint4*)(A + base) = *(const uint4*)o;
}

// ---- 2b) convert W [512,1024] f32 -> bf16 ----
__global__ __launch_bounds__(256) void cvt_bf16_kernel(const float* __restrict__ src,
                                                       unsigned short* __restrict__ dst, int n8) {
  int t = blockIdx.x * 256 + threadIdx.x;
  if (t >= n8) return;
  const float4* s = (const float4*)src + (size_t)t * 2;
  float4 lo = s[0], hi = s[1];
  alignas(16) unsigned short o[8] = { f2bf(lo.x), f2bf(lo.y), f2bf(lo.z), f2bf(lo.w),
                                      f2bf(hi.x), f2bf(hi.y), f2bf(hi.z), f2bf(hi.w) };
  *(uint4*)(dst + (size_t)t * 8) = *(const uint4*)o;
}

// ---- W panel chunk staging: 64 rows x BK cols into padded LDS slot ----
__device__ __forceinline__ void stage_chunk(const unsigned short* __restrict__ wpanel,
                                            unsigned short* __restrict__ dst,
                                            int kc, int tid) {
#pragma unroll
  for (int i = 0; i < 4; ++i) {                 // 64 * 128 ushort = 1024 uint4 / 256 thr
    int idx = tid + i * 256;
    int rp  = idx >> 4;                         // panel row (BK/8 = 16 uint4 per row)
    int cp  = (idx & 15) * 8;                   // ushort col within chunk
    const unsigned short* g = wpanel + (size_t)rp * K_DIM + kc + cp;
    unsigned short*       l = dst + rp * BSTR + cp;
#if USE_ASYNC_LDS
    // Async DMA: LDS byte offset in VDST (low 32 bits of the flat LDS address),
    // 64-bit global address in VADDR. Tracked by ASYNCcnt, no VGPR round-trip.
    unsigned           lds_addr = (unsigned)(size_t)(void*)l;
    unsigned long long gaddr    = (unsigned long long)(size_t)(const void*)g;
    asm volatile("global_load_async_to_lds_b128 %0, %1, off"
                 :: "v"(lds_addr), "v"(gaddr)
                 : "memory");
#else
    *(uint4*)l = *(const uint4*)g;
#endif
  }
}

__device__ __forceinline__ void wait_async_all() {
#if USE_ASYNC_LDS
  asm volatile("s_wait_asynccnt 0" ::: "memory");
#endif
}

// ---- 3) WMMA GEMM + bias + ReLU + BatchNorm epilogue ----
// Block = 256 threads (8 waves), tile 128(M) x 64(N); wave tile 32x32 (4 accumulators).
// W panel double-buffered in LDS via async global->LDS copies.
__global__ __launch_bounds__(256) void gemm_bn_kernel(
    const unsigned short* __restrict__ A,    // [N_NODES][K_DIM] bf16
    const unsigned short* __restrict__ Wb,   // [OUT_DIM][K_DIM] bf16 (row-major over K)
    const float* __restrict__ bias, const float* __restrict__ gamma,
    const float* __restrict__ beta, const float* __restrict__ mean,
    const float* __restrict__ var, float* __restrict__ out) {
  __shared__ unsigned short Bs[2][64 * BSTR];     // 2 x 17 KB double buffer

  const int tid  = threadIdx.x;
  const int bm   = blockIdx.x;        // 0..511
  const int bn   = blockIdx.y;        // 0..7
  const int wave = tid >> 5, lane = tid & 31;
  const int wm = wave & 3, wn = wave >> 2;        // 4 x 2 wave grid
  const int r = lane & 15, h = lane >> 4;         // CDNA5 16-bit fragment layout
  const int rowBase = bm * 128 + wm * 32;
  const int colBase = wn * 32;                    // within the 64-wide panel

  const unsigned short* wpanel = Wb + (size_t)bn * 64 * K_DIM;

  stage_chunk(wpanel, Bs[0], 0, tid);             // prologue: chunk 0
  wait_async_all();
  __syncthreads();

  v8f c00 = {}, c01 = {}, c10 = {}, c11 = {};

  int buf = 0;
  for (int kc = 0; kc < K_DIM; kc += BK, buf ^= 1) {
    if (kc + BK < K_DIM)                          // kick next chunk while computing
      stage_chunk(wpanel, Bs[buf ^ 1], kc + BK, tid);

    const unsigned short* base_a = A + (size_t)(rowBase + r) * K_DIM + kc + h * 8;
    const unsigned short* base_b = &Bs[buf][(colBase + r) * BSTR + h * 8];
    if (kc + BK < K_DIM) {                        // pull next A lines toward the WGP
      __builtin_prefetch(base_a + BK, 0, 0);
      __builtin_prefetch(base_a + (size_t)16 * K_DIM + BK, 0, 0);
    }

#pragma unroll
    for (int k0 = 0; k0 < BK; k0 += 32) {
      union U { uint4 u[2]; v16bf v; };
      U a0, a1, b0, b1;
      // A fragment: lane (r,h) holds K-chunks [h*8, h*8+8) and [16+h*8, 16+h*8+8)
      const unsigned short* ap = base_a + k0;
      a0.u[0] = *(const uint4*)(ap);
      a0.u[1] = *(const uint4*)(ap + 16);
      a1.u[0] = *(const uint4*)(ap + (size_t)16 * K_DIM);
      a1.u[1] = *(const uint4*)(ap + (size_t)16 * K_DIM + 16);
      // B fragment: same striping, panel row = output column
      const unsigned short* bp = base_b + k0;
      b0.u[0] = *(const uint4*)(bp);
      b0.u[1] = *(const uint4*)(bp + 16);
      b1.u[0] = *(const uint4*)(bp + 16 * BSTR);
      b1.u[1] = *(const uint4*)(bp + 16 * BSTR + 16);

      c00 = __builtin_amdgcn_wmma_f32_16x16x32_bf16(false, a0.v, false, b0.v, (short)0, c00, false, false);
      c01 = __builtin_amdgcn_wmma_f32_16x16x32_bf16(false, a0.v, false, b1.v, (short)0, c01, false, false);
      c10 = __builtin_amdgcn_wmma_f32_16x16x32_bf16(false, a1.v, false, b0.v, (short)0, c10, false, false);
      c11 = __builtin_amdgcn_wmma_f32_16x16x32_bf16(false, a1.v, false, b1.v, (short)0, c11, false, false);
    }

    wait_async_all();                             // next chunk landed in LDS
    __syncthreads();                              // everyone done reading cur chunk
  }

  // Epilogue. C layout: VGPR v, lane L -> row v + 8*(L>=16), col L&15.
  const int col0 = bn * 64 + colBase + r;
  const int col1 = col0 + 16;
  float s0 = gamma[col0] * rsqrtf(var[col0] + 1e-5f);
  float t0 = beta[col0] - mean[col0] * s0;
  float bb0 = bias[col0];
  float s1 = gamma[col1] * rsqrtf(var[col1] + 1e-5f);
  float t1 = beta[col1] - mean[col1] * s1;
  float bb1 = bias[col1];
#pragma unroll
  for (int v = 0; v < 8; ++v) {
    int row0 = rowBase + v + h * 8;
    int row1 = row0 + 16;
    float y;
    y = fmaxf(c00[v] + bb0, 0.f); out[(size_t)row0 * OUT_DIM + col0] = y * s0 + t0;
    y = fmaxf(c01[v] + bb1, 0.f); out[(size_t)row0 * OUT_DIM + col1] = y * s1 + t1;
    y = fmaxf(c10[v] + bb0, 0.f); out[(size_t)row1 * OUT_DIM + col0] = y * s0 + t0;
    y = fmaxf(c11[v] + bb1, 0.f); out[(size_t)row1 * OUT_DIM + col1] = y * s1 + t1;
  }
}

// ---- 4) row L2 normalize in place: one wave32 per row ----
__global__ __launch_bounds__(256) void l2norm_kernel(float* __restrict__ out) {
  unsigned rowi = (blockIdx.x * 256u + threadIdx.x) >> 5;
  unsigned lane = threadIdx.x & 31u;
  if (rowi >= N_NODES) return;
  float4* row = (float4*)(out + (size_t)rowi * OUT_DIM);
  float4 v[4];
  float s = 0.f;
#pragma unroll
  for (int i = 0; i < 4; ++i) {
    v[i] = row[lane + i * 32];
    s += v[i].x * v[i].x + v[i].y * v[i].y + v[i].z * v[i].z + v[i].w * v[i].w;
  }
#pragma unroll
  for (int off = 16; off > 0; off >>= 1) s += __shfl_down(s, off, 32);
  s = __shfl(s, 0, 32);
  float inv = 1.f / (sqrtf(s) + 1e-6f);
#pragma unroll
  for (int i = 0; i < 4; ++i) {
    float4 w = v[i];
    w.x *= inv; w.y *= inv; w.z *= inv; w.w *= inv;
    row[lane + i * 32] = w;
  }
}

extern "C" void kernel_launch(void* const* d_in, const int* in_sizes, int n_in,
                              void* d_out, int out_size, void* d_ws, size_t ws_size,
                              hipStream_t stream) {
  const float*      feat  = (const float*)d_in[0];
  const long long*  ei    = (const long long*)d_in[1];   // int64 edge_index [2, N_EDGES]
  const float*      W     = (const float*)d_in[2];
  const float*      bias  = (const float*)d_in[3];
  const float*      gamma = (const float*)d_in[4];
  const float*      beta  = (const float*)d_in[5];
  const float*      mean  = (const float*)d_in[6];
  const float*      var   = (const float*)d_in[7];
  float*            out   = (float*)d_out;

  char* ws = (char*)d_ws;
  float* agg = (float*)ws;                                            // 128 MB f32
  size_t aggBytes = (size_t)N_NODES * IN_DIM * sizeof(float);
  unsigned short* Abf = (unsigned short*)(ws + aggBytes);             // 128 MB bf16
  size_t aBytes = (size_t)N_NODES * K_DIM * sizeof(unsigned short);
  unsigned short* Wbf = (unsigned short*)(ws + aggBytes + aBytes);    // 1 MB bf16

  {
    int n4 = N_NODES * IN_DIM / 4;
    zero_f32_kernel<<<(n4 + 255) / 256, 256, 0, stream>>>((float4*)agg, n4);
  }
  scatter_kernel<<<N_EDGES * 32 / 256, 256, 0, stream>>>(feat, ei, agg);
  pack_A_kernel<<<(int)((size_t)N_NODES * K_DIM / 8 / 256), 256, 0, stream>>>(feat, agg, Abf);
  cvt_bf16_kernel<<<OUT_DIM * K_DIM / 8 / 256, 256, 0, stream>>>(W, Wbf, OUT_DIM * K_DIM / 8);

  dim3 g(N_NODES / 128, OUT_DIM / 64);
  gemm_bn_kernel<<<g, 256, 0, stream>>>(Abf, Wbf, bias, gamma, beta, mean, var, out);

  l2norm_kernel<<<N_NODES * 32 / 256, 256, 0, stream>>>(out);
}
